// GRUModel_21492016349847
// MI455X (gfx1250) — matrix operand order repaired and checked by reading, not compiled, not run on hardware
//
#include <hip/hip_runtime.h>
#include <hip/hip_bf16.h>

// ---------------------------------------------------------------------------
// CDNA5 (gfx1250) GRU stack: bf16 WMMA + TDM tensor loads + wave32 layout.
// ---------------------------------------------------------------------------

typedef __attribute__((ext_vector_type(16))) __bf16 v16bf;
typedef __attribute__((ext_vector_type(8)))  float  v8f;
typedef __attribute__((ext_vector_type(4)))  unsigned int u32x4;
typedef __attribute__((ext_vector_type(8)))  int    i32x8;
typedef __attribute__((ext_vector_type(4)))  int    i32x4;
typedef __bf16 bf16;

#define DEVINL __device__ __forceinline__

namespace {
constexpr int kB = 128, kT = 384, kF = 256, kG = 768, kU = 256;
constexpr int kNT   = kG / 16;        // 48 N-tiles of 16
constexpr int kKT   = kF / 32;        // 8 K-tiles of 32
constexpr int kRowT = kB * kT / 16;   // 3072 row tiles over B*T
constexpr int kFragE = 512;           // elems per 16x32 bf16 fragment tile (32 lanes * 16)
}

DEVINL v8f wmma_bf16(v16bf a, v16bf b, v8f c) {
  return __builtin_amdgcn_wmma_f32_16x16x32_bf16(false, a, false, b, (short)0, c,
                                                 false, false);
}

DEVINL unsigned lds_off_of(const void* p) {
  // Generic pointers to LDS carry the byte offset in the low 32 bits.
  return (unsigned)(unsigned long long)p;
}

// 1-D TDM copy global->LDS (nbytes multiple of 8, < 512KB). Issued by wave 0.
DEVINL void tdm_load_1d(unsigned lds_off, const void* gptr, unsigned nbytes) {
#if __has_builtin(__builtin_amdgcn_tensor_load_to_lds)
  unsigned long long ga = (unsigned long long)gptr;
  unsigned n8 = nbytes >> 3;                       // data_size = 8B units
  u32x4 g0;
  g0.x = 1u;                                       // count=1, user descriptor
  g0.y = lds_off;                                  // lds_addr
  g0.z = (unsigned)ga;                             // global_addr[31:0]
  g0.w = (unsigned)((ga >> 32) & 0x01FFFFFFu) | (2u << 30);  // addr hi | type=2
  i32x8 g1;
  g1[0] = (int)(3u << 16);                         // wg_mask=0, data_size=3 (8B)
  g1[1] = (int)((n8 & 0xFFFFu) << 16);             // tensor_dim0 lo16
  g1[2] = (int)((n8 >> 16) | (1u << 16));          // tensor_dim0 hi16 | tensor_dim1=1
  g1[3] = (int)((n8 & 0xFFFFu) << 16);             // tile_dim0
  g1[4] = 1;                                       // tile_dim1=1, tile_dim2=0
  g1[5] = (int)n8;                                 // tensor_dim0_stride lo32
  g1[6] = 0;
  g1[7] = 0;
  i32x4 g2 = {0, 0, 0, 0};
  i32x4 g3 = {0, 0, 0, 0};
  i32x8 g4 = {0, 0, 0, 0, 0, 0, 0, 0};             // unused trailing group (6-arg form)
  __builtin_amdgcn_tensor_load_to_lds(g0, g1, g2, g3, g4, 0);
#else
  (void)lds_off; (void)gptr; (void)nbytes;
#endif
}

DEVINL void load_tile_to_lds(void* lds_dst, const void* gsrc, unsigned nbytes,
                             int tid, int nthreads) {
#if __has_builtin(__builtin_amdgcn_tensor_load_to_lds)
  (void)nthreads;
  if (tid < 32) {
    tdm_load_1d(lds_off_of(lds_dst), gsrc, nbytes);
    __builtin_amdgcn_s_wait_tensorcnt(0);
  }
#else
  const uint4* s = (const uint4*)gsrc;
  uint4* d = (uint4*)lds_dst;
  for (unsigned i = tid; i < nbytes / 16; i += nthreads) d[i] = s[i];
#endif
}

// ---------------------------------------------------------------------------
// Packing kernels: swizzle fp32 inputs into bf16 WMMA fragment-major layouts.
// A-layout (16x32 bf16): lane holds row=lane&15; K = (lane>>4)*8 + (e&7) + (e>=8)*16
// B-layout (32x16 bf16): lane holds col=lane&15; K = (lane>>4)*8 + (e&7) + (e>=8)*16
// ---------------------------------------------------------------------------

__global__ void pack_weights_kernel(const float* __restrict__ src,  // [256, 768]
                                    bf16* __restrict__ dst) {       // [48][8][512]
  int idx = blockIdx.x * 256 + threadIdx.x;
  if (idx >= kNT * kKT * kFragE) return;
  int e = idx & 15, lane = (idx >> 4) & 31, kt = (idx >> 9) & 7, nt = idx >> 12;
  int K = kt * 32 + (lane >> 4) * 8 + (e & 7) + ((e >> 3) << 4);
  int N = nt * 16 + (lane & 15);
  dst[idx] = (bf16)src[K * kG + N];
}

__global__ void pack_x_kernel(const float* __restrict__ x,   // [B*T, 256]
                              bf16* __restrict__ dst) {      // [3072][8][512]
  int idx = blockIdx.x * 256 + threadIdx.x;
  if (idx >= kRowT * kKT * kFragE) return;
  int e = idx & 15, lane = (idx >> 4) & 31, kt = (idx >> 9) & 7, rt = idx >> 12;
  int row = rt * 16 + (lane & 15);
  int K = kt * 32 + (lane >> 4) * 8 + (e & 7) + ((e >> 3) << 4);
  dst[idx] = (bf16)x[(size_t)row * kF + K];
}

// ---------------------------------------------------------------------------
// Input projection: XP[B*T,768] = A[B*T,256] @ W[256,768] + bias, bf16 out.
// Grid (768, 3): each WG does 64 rows x 256 cols, K=256 in one shot.
// LDS: A 32KB + B 128KB, both via one TDM load each.
// ---------------------------------------------------------------------------

__global__ void __launch_bounds__(256, 1)
proj_kernel(const bf16* __restrict__ Afrag,   // [3072][8][512] A-frag major
            const bf16* __restrict__ Bfrag,   // [48][8][512]  B-frag major
            const float* __restrict__ bias_in,// [768]
            bf16* __restrict__ xp) {          // [B*T][768]
  extern __shared__ char smem[];
  bf16* sA = (bf16*)smem;                 // 4 rowtiles * 8 kt * 512 = 32KB
  bf16* sB = (bf16*)(smem + 32768);       // 16 ntiles  * 8 kt * 512 = 128KB
  const int rb = blockIdx.x;              // row-tile block (4 tiles)
  const int cb = blockIdx.y;              // 256-col block
  const int tid = threadIdx.x, lane = tid & 31, w = tid >> 5;

  load_tile_to_lds(sA, Afrag + (size_t)rb * 4 * kKT * kFragE, 32768, tid, 256);
  load_tile_to_lds(sB, Bfrag + (size_t)cb * 16 * kKT * kFragE, 131072, tid, 256);
  __syncthreads();

  v8f acc[4][2];
#pragma unroll
  for (int mt = 0; mt < 4; ++mt)
#pragma unroll
    for (int j = 0; j < 2; ++j) acc[mt][j] = (v8f){0, 0, 0, 0, 0, 0, 0, 0};

#pragma unroll
  for (int kt = 0; kt < kKT; ++kt) {
    v16bf b0 = *(const v16bf*)(sB + ((2 * w + 0) * kKT + kt) * kFragE + lane * 16);
    v16bf b1 = *(const v16bf*)(sB + ((2 * w + 1) * kKT + kt) * kFragE + lane * 16);
#pragma unroll
    for (int mt = 0; mt < 4; ++mt) {
      v16bf a = *(const v16bf*)(sA + (mt * kKT + kt) * kFragE + lane * 16);
      acc[mt][0] = wmma_bf16(a, b0, acc[mt][0]);
      acc[mt][1] = wmma_bf16(a, b1, acc[mt][1]);
    }
  }

#pragma unroll
  for (int mt = 0; mt < 4; ++mt) {
#pragma unroll
    for (int j = 0; j < 2; ++j) {
      int nt = 2 * w + j;
      int col = cb * 256 + nt * 16 + (lane & 15);
      float bi = bias_in[col];
#pragma unroll
      for (int i = 0; i < 8; ++i) {
        int row = (rb * 4 + mt) * 16 + (lane >> 4) * 8 + i;
        xp[(size_t)row * kG + col] = (bf16)(acc[mt][j][i] + bi);
      }
    }
  }
}

// ---------------------------------------------------------------------------
// Recurrence: 8 persistent WGs, one 16-batch slice each. z/r-gate weights
// (262KB) LDS-resident via TDM; h-gate weights (16 tiles) VGPR-resident.
// Wave w owns units [32w, 32w+32): all 3 gates local to the wave.
// ---------------------------------------------------------------------------

__global__ void __launch_bounds__(256, 1)
gru_rec_kernel(const bf16* __restrict__ xp,     // [(b*T+t)][768] bf16 (incl. input bias)
               const bf16* __restrict__ Rfrag,  // [48][8][512] B-frag major
               const float* __restrict__ brec,  // [768] recurrent bias
               bf16* __restrict__ hseq,         // A-frag-major output (flags&1)
               float* __restrict__ hout,        // [B,256] final state (flags&2)
               int flags) {
  extern __shared__ char smem[];
  bf16* sRk = (bf16*)smem;                   // n-tiles 0..31 (z,r gates): 262144 B
  bf16* sH  = (bf16*)(smem + 262144);        // h fragment buffer: 8*512 bf16 = 8192 B
  const int tid = threadIdx.x, lane = tid & 31, w = tid >> 5;
  const int b0 = blockIdx.x * 16;

  load_tile_to_lds(sRk, Rfrag, 262144, tid, 256);
  for (int i = tid; i < kKT * kFragE; i += 256) sH[i] = (bf16)0.0f;

  // VGPR-resident h-gate weights: global n-tiles 32 + {2w, 2w+1}.
  v16bf rhw[2][8];
#pragma unroll
  for (int j = 0; j < 2; ++j)
#pragma unroll
    for (int kt = 0; kt < 8; ++kt)
      rhw[j][kt] = *(const v16bf*)(Rfrag +
          (size_t)(((32 + 2 * w + j) * kKT + kt) * kFragE + lane * 16));

  float bz[2], br_[2], bh[2];
#pragma unroll
  for (int j = 0; j < 2; ++j) {
    int u = 32 * w + j * 16 + (lane & 15);
    bz[j] = brec[u];
    br_[j] = brec[256 + u];
    bh[j] = brec[512 + u];
  }
  __syncthreads();

  const int writeSeq = flags & 1, writeFin = flags & 2;

  for (int t = 0; t < kT; ++t) {
    // xp gathers are independent of h: issue first, hide under WMMAs.
    float xz[2][8], xr[2][8], xh[2][8];
#pragma unroll
    for (int j = 0; j < 2; ++j) {
      int u = 32 * w + j * 16 + (lane & 15);
#pragma unroll
      for (int i = 0; i < 8; ++i) {
        int bb = b0 + (lane >> 4) * 8 + i;
        size_t base = ((size_t)bb * kT + t) * kG;
        xz[j][i] = (float)xp[base + u];
        xr[j][i] = (float)xp[base + 256 + u];
        xh[j][i] = (float)xp[base + 512 + u];
      }
    }

    v8f az[2], ar[2], ah[2];
#pragma unroll
    for (int j = 0; j < 2; ++j) {
      az[j] = (v8f){0, 0, 0, 0, 0, 0, 0, 0};
      ar[j] = (v8f){0, 0, 0, 0, 0, 0, 0, 0};
      ah[j] = (v8f){0, 0, 0, 0, 0, 0, 0, 0};
    }
#pragma unroll
    for (int kt = 0; kt < kKT; ++kt) {
      v16bf a = *(const v16bf*)(sH + kt * kFragE + lane * 16);
#pragma unroll
      for (int j = 0; j < 2; ++j) {
        v16bf wz = *(const v16bf*)(sRk + ((2 * w + j) * kKT + kt) * kFragE + lane * 16);
        v16bf wr = *(const v16bf*)(sRk + (((16 + 2 * w + j) * kKT + kt) * kFragE) + lane * 16);
        az[j] = wmma_bf16(a, wz, az[j]);
        ar[j] = wmma_bf16(a, wr, ar[j]);
        ah[j] = wmma_bf16(a, rhw[j][kt], ah[j]);
      }
    }

    float hnew[2][8];
#pragma unroll
    for (int j = 0; j < 2; ++j) {
      int u = 32 * w + j * 16 + (lane & 15);
      int kt2 = u >> 5, K = u & 31;
      int half = (K >> 3) & 1, e = (K & 7) | ((K >> 4) << 3);
#pragma unroll
      for (int i = 0; i < 8; ++i) {
        int row = (lane >> 4) * 8 + i;
        float hold = (float)sH[(kt2 * 32 + half * 16 + row) * 16 + e];
        float z = 1.0f / (1.0f + __expf(-(xz[j][i] + az[j][i] + bz[j])));
        float r = 1.0f / (1.0f + __expf(-(xr[j][i] + ar[j][i] + br_[j])));
        float hh = tanhf(xh[j][i] + r * (ah[j][i] + bh[j]));
        hnew[j][i] = z * hold + (1.0f - z) * hh;
      }
    }
    __syncthreads();  // everyone done reading old h

#pragma unroll
    for (int j = 0; j < 2; ++j) {
      int u = 32 * w + j * 16 + (lane & 15);
      int kt2 = u >> 5, K = u & 31;
      int half = (K >> 3) & 1, e = (K & 7) | ((K >> 4) << 3);
#pragma unroll
      for (int i = 0; i < 8; ++i) {
        int row = (lane >> 4) * 8 + i;
        bf16 hb = (bf16)hnew[j][i];
        sH[(kt2 * 32 + half * 16 + row) * 16 + e] = hb;
        if (writeSeq) {
          int bb = b0 + row;
          int rt = bb * (kT / 16) + (t >> 4);
          int rr = t & 15;
          hseq[(size_t)((rt * kKT + kt2) * 32 + half * 16 + rr) * 16 + e] = hb;
        }
        if (writeFin && t == kT - 1) {
          hout[(size_t)(b0 + row) * kU + u] = hnew[j][i];
        }
      }
    }
    __syncthreads();  // new h visible for next step's WMMAs
  }
}

// ---------------------------------------------------------------------------
// Host side
// ---------------------------------------------------------------------------

extern "C" void kernel_launch(void* const* d_in, const int* in_sizes, int n_in,
                              void* d_out, int out_size, void* d_ws, size_t ws_size,
                              hipStream_t stream) {
  (void)in_sizes; (void)n_in; (void)out_size; (void)ws_size;
  const float* x     = (const float*)d_in[0];  // [128,384,256]
  const float* k0    = (const float*)d_in[1];  // [256,768]
  const float* rk0   = (const float*)d_in[2];  // [256,768]
  const float* b0_   = (const float*)d_in[3];  // [2,768]
  const float* kern  = (const float*)d_in[4];  // [3,256,768]
  const float* rkern = (const float*)d_in[5];  // [3,256,768]
  const float* bias  = (const float*)d_in[6];  // [3,2,768]
  float* out = (float*)d_out;

  char* ws = (char*)d_ws;
  const size_t szXb  = (size_t)kRowT * kKT * kFragE * 2;   // 25,165,824
  const size_t szW   = (size_t)kNT * kKT * kFragE * 2;     //    393,216
  const size_t szXp  = (size_t)kB * kT * kG * 2;           // 75,497,472
  bf16* xb    = (bf16*)(ws);
  bf16* hseqA = (bf16*)(ws + szXb);
  bf16* hseqB = (bf16*)(ws + 2 * szXb);
  bf16* xp    = (bf16*)(ws + 3 * szXb);
  bf16* wk[4];
  bf16* wr[4];
  for (int L = 0; L < 4; ++L) {
    wk[L] = (bf16*)(ws + 3 * szXb + szXp + (size_t)L * szW);
    wr[L] = (bf16*)(ws + 3 * szXb + szXp + 4 * szW + (size_t)L * szW);
  }

  // --- pack inputs/weights into WMMA fragment layouts (bf16) ---
  pack_x_kernel<<<kRowT * kKT * kFragE / 256, 256, 0, stream>>>(x, xb);
  const int wgrid = kNT * kKT * kFragE / 256;  // 768
  pack_weights_kernel<<<wgrid, 256, 0, stream>>>(k0, wk[0]);
  pack_weights_kernel<<<wgrid, 256, 0, stream>>>(rk0, wr[0]);
  for (int L = 1; L < 4; ++L) {
    pack_weights_kernel<<<wgrid, 256, 0, stream>>>(kern + (size_t)(L - 1) * kF * kG, wk[L]);
    pack_weights_kernel<<<wgrid, 256, 0, stream>>>(rkern + (size_t)(L - 1) * kF * kG, wr[L]);
  }

  const bf16* Ain = xb;
  bf16* seqs[4] = {hseqA, hseqB, hseqA, hseqB /*unused*/};
  for (int L = 0; L < 4; ++L) {
    const float* bin = (L == 0) ? b0_ : (bias + (size_t)(L - 1) * 2 * kG);
    const float* brc = bin + kG;
    proj_kernel<<<dim3(kRowT / 4, 3), 256, 163840, stream>>>(Ain, wk[L], bin, xp);
    int flags = (L < 3 ? 1 : 0) | (L == 3 ? 2 : 0);
    gru_rec_kernel<<<kB / 16, 256, 270336, stream>>>(xp, wr[L], brc, seqs[L], out, flags);
    Ain = seqs[L];
  }
}
